// AdapterLayer_27771258536606
// MI455X (gfx1250) — matrix operand order repaired and checked by reading, not compiled, not run on hardware
//
#include <hip/hip_runtime.h>
#include <cstdint>
#include <cstddef>

// ---------------------------------------------------------------------------
// MI455X (gfx1250) fused pre-norm adapter:
//   out = x + (gelu(LN(x) @ Wd + bd) @ Wu + bu)
// B*S = 32768 tokens, H = 1024, R = 64, fp32 in/out.
//
// Roofline: ~8.6 GFLOP vs ~268 MB HBM => memory bound at 23.3 TB/s (~11.5us).
// f16 WMMA (v_wmma_f32_16x16x32_f16, f32 accum) keeps compute ~3x under the
// memory floor; fp32 WMMA (16x16x4) would not. LN/GELU/residual stay fp32.
// Weights pre-swizzled to f16 WMMA B-fragment layout in d_ws (L2-resident);
// up-proj fragments are staged LDS-side via GLOBAL_LOAD_ASYNC_TO_LDS_B128
// issued at block start and settled (s_wait_asynccnt) only before phase 4,
// fully overlapping the copy with LN + down-proj + GELU.
// ---------------------------------------------------------------------------

typedef __attribute__((ext_vector_type(16))) _Float16 v16h;
typedef __attribute__((ext_vector_type(8)))  _Float16 v8h;
typedef __attribute__((ext_vector_type(8)))  float    v8f;
typedef __attribute__((ext_vector_type(4)))  float    v4f;

namespace {
constexpr int   kH        = 1024;
constexpr int   kR        = 64;
constexpr float kLnEps    = 1e-5f;
constexpr int   kDnKSteps = kH / 32;   // 32 K-steps for down-proj
constexpr int   kDnNT     = kR / 16;   // 4  N-tiles for down-proj
constexpr int   kUpNT     = kH / 16;   // 64 N-tiles for up-proj
constexpr int   kDnFragHalves = kDnNT * kDnKSteps * 32 * 16;  // 65536 (128 KB)
constexpr int   kUpFragHalves = kUpNT * 2 * 32 * 16;          // 65536 (128 KB)
constexpr int   kActStride = kR + 8;   // padded LDS row to spread banks
}  // namespace

// ---------------------------------------------------------------------------
// Pre-swizzle weights into f16 WMMA B fragments.
// B-frag k order used here, A matches (16x16x32 f16): lane n<16 holds K=0..15 (halves 0..15),
// lane 16+n holds K=16..31; column index = lane & 15.
// dn frag element addr: ((t*32 + s)*32 + lane)*16 + j, k = s*32+(lane>>4)*16+j
// up frag element addr: ((t2*2 + s)*32 + lane)*16 + j, k = s*32+(lane>>4)*16+j
// ---------------------------------------------------------------------------
__global__ void convert_weights_kernel(const float* __restrict__ w_down,
                                       const float* __restrict__ w_up,
                                       _Float16* __restrict__ frags) {
  const int idx = blockIdx.x * blockDim.x + threadIdx.x;
  if (idx < kDnFragHalves) {
    const int j    = idx & 15;
    const int lane = (idx >> 4) & 31;
    const int s    = (idx >> 9) & 31;
    const int t    = idx >> 14;
    const int n    = lane & 15;
    const int k    = s * 32 + ((lane >> 4) << 4) + j;
    frags[idx] = (_Float16)w_down[k * kR + t * 16 + n];
  } else if (idx < kDnFragHalves + kUpFragHalves) {
    const int i2   = idx - kDnFragHalves;
    const int j    = i2 & 15;
    const int lane = (i2 >> 4) & 31;
    const int s    = (i2 >> 9) & 1;
    const int t    = i2 >> 10;
    const int n    = lane & 15;
    const int k    = s * 32 + ((lane >> 4) << 4) + j;
    frags[idx] = (_Float16)w_up[k * kH + t * 16 + n];
  }
}

__device__ __forceinline__ float gelu_erf(float v) {
  return 0.5f * v * (1.0f + erff(v * 0.70710678118654752f));
}

__device__ __forceinline__ v8f wmma_f16(v16h a, v16h b, v8f c) {
  return __builtin_amdgcn_wmma_f32_16x16x32_f16(false, a, false, b,
                                                (short)0, c, false, false);
}

// ---------------------------------------------------------------------------
// 256 threads = 8 wave32s; each wave owns one 16-token tile.
// Dynamic LDS: 128 KB of up-proj B fragments (async-copied at block start).
// ---------------------------------------------------------------------------
__global__ __launch_bounds__(256) void adapter_fused_kernel(
    const float* __restrict__ x,
    const float* __restrict__ ln_gamma,
    const float* __restrict__ ln_beta,
    const float* __restrict__ b_down,
    const float* __restrict__ b_up,
    const _Float16* __restrict__ dn_frags,
    const _Float16* __restrict__ up_frags,
    float* __restrict__ out) {
  extern __shared__ __align__(16) _Float16 s_up[];  // kUpFragHalves (128 KB)
  __shared__ float s_gamma[kH];
  __shared__ float s_beta[kH];
  __shared__ __align__(32) _Float16 s_act[8][16 * kActStride];  // gelu scratch

  const int tid = threadIdx.x;

  // ---- Kick off async staging of up-proj fragments into LDS. Tracked on
  // ASYNCcnt; consumed only in phase 4, so the DMA overlaps phases 1-3. ----
  {
    const uint32_t lds_base = (uint32_t)(uintptr_t)&s_up[0];  // low 32b = LDS off
    for (int i = tid; i < kUpFragHalves / 8; i += 256) {      // 16 B per copy
      const void* g = (const void*)(up_frags + (size_t)i * 8);
      const uint32_t l = lds_base + (uint32_t)i * 16u;
      asm volatile("global_load_async_to_lds_b128 %0, %1, off"
                   :: "v"(l), "v"(g) : "memory");
    }
  }

  {  // stage gamma/beta in LDS (256 threads * 4 floats = 1024)
    const int i = tid * 4;
    *(v4f*)&s_gamma[i] = *(const v4f*)&ln_gamma[i];
    *(v4f*)&s_beta[i]  = *(const v4f*)&ln_beta[i];
  }
  __syncthreads();

  const int wave = tid >> 5;
  const int lane = tid & 31;
  const int hf   = lane >> 4;   // half-wave id (0/1)
  const int r    = lane & 15;   // row within 16-token tile / column within N-tile
  const int tile = blockIdx.x * 8 + wave;
  const float* __restrict__ xrow = x + (size_t)(tile * 16 + r) * kH;

  // ---- Phase 1: LayerNorm statistics (lane pair l, l^16 covers one row) ----
  float sum = 0.0f, sq = 0.0f;
  {
    const float* p = xrow + hf * (kH / 2);
#pragma unroll 4
    for (int i = 0; i < kH / 2; i += 4) {
      const v4f v = *(const v4f*)(p + i);
      sum += (v.x + v.y) + (v.z + v.w);
      sq = fmaf(v.x, v.x, sq); sq = fmaf(v.y, v.y, sq);
      sq = fmaf(v.z, v.z, sq); sq = fmaf(v.w, v.w, sq);
    }
  }
  sum += __shfl_xor(sum, 16, 32);
  sq  += __shfl_xor(sq, 16, 32);
  const float mu = sum * (1.0f / kH);
  const float rs = rsqrtf(fmaf(-mu, mu, sq * (1.0f / kH)) + kLnEps);

  // ---- Phase 2: down-proj, M=16 N=64 K=1024 via f16 WMMA, f32 accum ----
  // A-frag layout: lane m<16: halves 0..15 => K=kk+0..15;
  //                lane 16+m: halves 0..15 => K=kk+16..31.
  v8f acc0 = {}, acc1 = {}, acc2 = {}, acc3 = {};
  const int c0 = hf * 16;
  for (int ks = 0; ks < kDnKSteps; ++ks) {
    const int kk = ks * 32;
    if (ks + 1 < kDnKSteps) __builtin_prefetch(xrow + kk + 32, 0, 0);
    const v4f xa = *(const v4f*)(xrow + kk + c0);
    const v4f xb = *(const v4f*)(xrow + kk + c0 + 4);
    const v4f xc = *(const v4f*)(xrow + kk + 8 + c0);
    const v4f xd = *(const v4f*)(xrow + kk + 8 + c0 + 4);
    const v4f ga = *(const v4f*)&s_gamma[kk + c0];
    const v4f gb = *(const v4f*)&s_gamma[kk + c0 + 4];
    const v4f gc = *(const v4f*)&s_gamma[kk + 8 + c0];
    const v4f gd = *(const v4f*)&s_gamma[kk + 8 + c0 + 4];
    const v4f ba = *(const v4f*)&s_beta[kk + c0];
    const v4f bb = *(const v4f*)&s_beta[kk + c0 + 4];
    const v4f bc = *(const v4f*)&s_beta[kk + 8 + c0];
    const v4f bd = *(const v4f*)&s_beta[kk + 8 + c0 + 4];

    union { v16h v; _Float16 e[16]; } a;
#pragma unroll
    for (int q = 0; q < 4; ++q) {
      a.e[q]      = (_Float16)fmaf((xa[q] - mu) * rs, ga[q], ba[q]);
      a.e[4 + q]  = (_Float16)fmaf((xb[q] - mu) * rs, gb[q], bb[q]);
      a.e[8 + q]  = (_Float16)fmaf((xc[q] - mu) * rs, gc[q], bc[q]);
      a.e[12 + q] = (_Float16)fmaf((xd[q] - mu) * rs, gd[q], bd[q]);
    }

    const _Float16* fb = dn_frags + ((size_t)ks * 32 + lane) * 16;  // t = 0
    const v16h b0 = *(const v16h*)(fb);
    const v16h b1 = *(const v16h*)(fb + 16384);   // t-stride = 32*32*16 halves
    const v16h b2 = *(const v16h*)(fb + 32768);
    const v16h b3 = *(const v16h*)(fb + 49152);
    acc0 = wmma_f16(a.v, b0, acc0);
    acc1 = wmma_f16(a.v, b1, acc1);
    acc2 = wmma_f16(a.v, b2, acc2);
    acc3 = wmma_f16(a.v, b3, acc3);
  }

  // ---- Phase 3: +b_down, exact-erf GELU, transpose 16x64 through LDS ----
  // C/D layout: lane l, vgpr v => M = v + 8*(l>>4), N = l&15.
  _Float16* sm = s_act[wave];
  {
    const v8f accs[4] = {acc0, acc1, acc2, acc3};
#pragma unroll
    for (int t = 0; t < 4; ++t) {
      const float bdn = b_down[t * 16 + r];
#pragma unroll
      for (int v = 0; v < 8; ++v) {
        const float g = gelu_erf(accs[t][v] + bdn);
        sm[(v + 8 * hf) * kActStride + t * 16 + r] = (_Float16)g;
      }
    }
  }
  // Settle the async fragment DMA, then one barrier covers both the act
  // transpose (cross-lane LDS) and the staged up-proj fragments.
  asm volatile("s_wait_asynccnt 0x0" ::: "memory");
  __syncthreads();

  // ---- Phase 4: up-proj, M=16 N=1024 K=64; fuse +b_up and +residual ----
  const v8h lo0 = *(const v8h*)&sm[r * kActStride + c0];
  const v8h hi0 = *(const v8h*)&sm[r * kActStride + 8 + c0];
  const v8h lo1 = *(const v8h*)&sm[r * kActStride + 32 + c0];
  const v8h hi1 = *(const v8h*)&sm[r * kActStride + 40 + c0];
  const v16h a0 = __builtin_shufflevector(lo0, hi0, 0, 1, 2, 3, 4, 5, 6, 7,
                                          8, 9, 10, 11, 12, 13, 14, 15);
  const v16h a1 = __builtin_shufflevector(lo1, hi1, 0, 1, 2, 3, 4, 5, 6, 7,
                                          8, 9, 10, 11, 12, 13, 14, 15);

  const size_t rowbase = (size_t)tile * 16 * kH;
  for (int t2 = 0; t2 < kUpNT; ++t2) {
    const _Float16* fu = &s_up[((size_t)t2 * 2 * 32 + lane) * 16];
    const v16h b0 = *(const v16h*)fu;
    const v16h b1 = *(const v16h*)(fu + 512);  // s-stride = 32*16 halves
    v8f c = {};
    c = wmma_f16(a0, b0, c);
    c = wmma_f16(a1, b1, c);
    const int col = t2 * 16 + r;
    const float bu = b_up[col];
#pragma unroll
    for (int v = 0; v < 8; ++v) {
      const size_t off = rowbase + (size_t)(v + 8 * hf) * kH + col;
      out[off] = (c[v] + bu) + x[off];
    }
  }
}

extern "C" void kernel_launch(void* const* d_in, const int* in_sizes, int n_in,
                              void* d_out, int out_size, void* d_ws, size_t ws_size,
                              hipStream_t stream) {
  const float* x      = (const float*)d_in[0];
  const float* gamma  = (const float*)d_in[1];
  const float* beta   = (const float*)d_in[2];
  const float* w_down = (const float*)d_in[3];
  const float* b_down = (const float*)d_in[4];
  const float* w_up   = (const float*)d_in[5];
  const float* b_up   = (const float*)d_in[6];
  float* out = (float*)d_out;
  _Float16* frags = (_Float16*)d_ws;  // 256 KB of f16 fragments

  const int total_frags = kDnFragHalves + kUpFragHalves;
  convert_weights_kernel<<<(total_frags + 255) / 256, 256, 0, stream>>>(
      w_down, w_up, frags);

  const int ntok    = in_sizes[0] / kH;  // 32768
  const int nblocks = ntok / 128;        // 8 waves/block * 16 tokens/wave
  const size_t dyn_lds = (size_t)kUpFragHalves * sizeof(_Float16);  // 128 KB
  adapter_fused_kernel<<<nblocks, 256, dyn_lds, stream>>>(
      x, gamma, beta, b_down, b_up, frags, frags + kDnFragHalves, out);
}
